// CrossAttention_27049704030419
// MI455X (gfx1250) — matrix operand order repaired and checked by reading, not compiled, not run on hardware
//
#include <hip/hip_runtime.h>
#include <math.h>

#define DIM    1024
#define NHEADS 8
#define HDIM   128
#define BATCH  4
#define TSEQ   2048
#define KTOT   (TSEQ + 2)   // 2050 real keys (tokens + h + p)
#define TKP    2112         // padded to 33 tiles of 64 keys

typedef __attribute__((ext_vector_type(16))) _Float16 v16h;
typedef __attribute__((ext_vector_type(8)))  _Float16 v8h;
typedef __attribute__((ext_vector_type(8)))  float    v8f;

static __device__ __forceinline__ v16h cat16(v8h lo, v8h hi) {
  v16h r;
#pragma unroll
  for (int i = 0; i < 8; ++i) { r[i] = lo[i]; r[i + 8] = hi[i]; }
  return r;
}

// ---------------- elementwise f32 -> f16 ----------------
__global__ void cvt_f16(const float* __restrict__ in, _Float16* __restrict__ out, int n) {
  int i = blockIdx.x * blockDim.x + threadIdx.x;
  if (i < n) out[i] = (_Float16)in[i];
}

// ---------------- tiled WMMA GEMM: C = (A @ W^T + bias) * alpha ----------------
// A: (M,K) f16 row-major, W: (N,K) f16 row-major. Double-buffered LDS, 1 barrier/step.
// MODE 0: f16 out, row remapped (rowsInB rows/batch in -> rowsOutB rows/batch out)
// MODE 1: f32 out (final projection)
// MODE 2: f16 out transposed per-head: Vt[b][h][dh][t]
template <int MODE>
__global__ __launch_bounds__(256) void gemm_xwT(
    const _Float16* __restrict__ A, const _Float16* __restrict__ W,
    const float* __restrict__ bias, void* __restrict__ Cout,
    int M, int N, int K, float alpha, int rowsInB, int rowsOutB) {
  __shared__ _Float16 lA[2][128 * 40];  // +8 halves pad/row keeps 16B alignment
  __shared__ _Float16 lB[2][128 * 40];
  const int tid  = threadIdx.x;
  const int lane = tid & 31, wave = tid >> 5;
  const int li = lane & 15, lh = lane >> 4;
  const int wm = wave >> 2, wn = wave & 3;  // 2x4 waves -> 64x32 per wave
  const int m0 = blockIdx.y * 128, n0 = blockIdx.x * 128;

  // each thread stages 2 chunks of 8 halves for A and B
  const int ch0 = tid, ch1 = tid + 256;
  const int rA0 = ch0 >> 2, cA0 = (ch0 & 3) * 8;
  const int rA1 = ch1 >> 2, cA1 = (ch1 & 3) * 8;

  const v8f vzero = {0.f, 0.f, 0.f, 0.f, 0.f, 0.f, 0.f, 0.f};
  v8f acc[4][2];
#pragma unroll
  for (int i = 0; i < 4; ++i)
#pragma unroll
    for (int j = 0; j < 2; ++j) acc[i][j] = vzero;

  // prologue: fill buffer 0
  {
    *(v8h*)&lA[0][rA0 * 40 + cA0] = *(const v8h*)&A[(size_t)(m0 + rA0) * K + cA0];
    *(v8h*)&lA[0][rA1 * 40 + cA1] = *(const v8h*)&A[(size_t)(m0 + rA1) * K + cA1];
    *(v8h*)&lB[0][rA0 * 40 + cA0] = *(const v8h*)&W[(size_t)(n0 + rA0) * K + cA0];
    *(v8h*)&lB[0][rA1 * 40 + cA1] = *(const v8h*)&W[(size_t)(n0 + rA1) * K + cA1];
  }
  __syncthreads();

  int cur = 0;
  for (int k0 = 0; k0 < K; k0 += 32) {
    // issue next tile's global loads (overlap with WMMAs below)
    v8h sa0, sa1, sb0, sb1;
    const bool more = (k0 + 32) < K;
    if (more) {
      int kn = k0 + 32;
      sa0 = *(const v8h*)&A[(size_t)(m0 + rA0) * K + kn + cA0];
      sa1 = *(const v8h*)&A[(size_t)(m0 + rA1) * K + kn + cA1];
      sb0 = *(const v8h*)&W[(size_t)(n0 + rA0) * K + kn + cA0];
      sb1 = *(const v8h*)&W[(size_t)(n0 + rA1) * K + kn + cA1];
    }

    v16h af[4], bf[2];
#pragma unroll
    for (int i = 0; i < 4; ++i) {  // A frag: lane li = row; halves {0-7,16-23}|{8-15,24-31}
      int r = wm * 64 + i * 16 + li;
      const _Float16* p = &lA[cur][r * 40 + lh * 8];
      af[i] = cat16(*(const v8h*)p, *(const v8h*)(p + 16));
    }
#pragma unroll
    for (int j = 0; j < 2; ++j) {  // B frag: lane li = column; halves {0-7,16-23}|{8-15,24-31}
      int cL = wn * 32 + j * 16 + li;
      const _Float16* p = &lB[cur][cL * 40 + lh * 8];
      bf[j] = cat16(*(const v8h*)p, *(const v8h*)(p + 16));
    }
#pragma unroll
    for (int i = 0; i < 4; ++i)
#pragma unroll
      for (int j = 0; j < 2; ++j)
        acc[i][j] = __builtin_amdgcn_wmma_f32_16x16x32_f16(
            false, af[i], false, bf[j], (short)0, acc[i][j], false, false);

    if (more) {  // stash into the other buffer (no conflict with buf[cur] readers)
      int nxt = cur ^ 1;
      *(v8h*)&lA[nxt][rA0 * 40 + cA0] = sa0;
      *(v8h*)&lA[nxt][rA1 * 40 + cA1] = sa1;
      *(v8h*)&lB[nxt][rA0 * 40 + cA0] = sb0;
      *(v8h*)&lB[nxt][rA1 * 40 + cA1] = sb1;
    }
    __syncthreads();
    cur ^= 1;
  }

  // epilogue: C layout VGPR v -> row v + 8*lh, lane li -> col
#pragma unroll
  for (int i = 0; i < 4; ++i)
#pragma unroll
    for (int j = 0; j < 2; ++j)
#pragma unroll
      for (int v = 0; v < 8; ++v) {
        int row = m0 + wm * 64 + i * 16 + v + 8 * lh;
        int col = n0 + wn * 32 + j * 16 + li;
        float val = (acc[i][j][v] + bias[col]) * alpha;
        if (MODE == 1) {
          ((float*)Cout)[(size_t)row * N + col] = val;
        } else if (MODE == 0) {
          int b = row / rowsInB, t = row - b * rowsInB;
          ((_Float16*)Cout)[((size_t)b * rowsOutB + t) * N + col] = (_Float16)val;
        } else {
          int b = row / rowsInB, t = row - b * rowsInB;
          int hh = col >> 7, dh = col & 127;
          ((_Float16*)Cout)[(((size_t)b * NHEADS + hh) * HDIM + dh) * TKP + t] = (_Float16)val;
        }
      }
}

// ---------------- adapter rows: (hp @ W^T + b) [* tanh(g)] ----------------
__global__ void adapter_kv(const float* __restrict__ hb, const float* __restrict__ pb,
                           const float* __restrict__ W, const float* __restrict__ bias,
                           const float* __restrict__ g, _Float16* __restrict__ out, int mode) {
  int idx = blockIdx.x * blockDim.x + threadIdx.x;  // B*2*DIM outputs
  if (idx >= BATCH * 2 * DIM) return;
  int n = idx & (DIM - 1);
  int j = (idx >> 10) & 1;
  int b = idx >> 11;
  const float* src = (j ? pb : hb) + (size_t)b * DIM;
  float acc = bias[n];
  for (int k = 0; k < DIM; ++k) acc += src[k] * W[(size_t)n * DIM + k];
  if (mode == 0) {  // K adapter rows, gate folded in
    acc *= tanhf(g[0]);
    out[((size_t)b * TKP + TSEQ + j) * DIM + n] = (_Float16)acc;
  } else {          // V adapter columns of transposed Vt
    out[(((size_t)b * NHEADS + (n >> 7)) * HDIM + (n & 127)) * TKP + (TSEQ + j)] = (_Float16)acc;
  }
}

// ---------------- flash attention: one wave = 16 Q rows, 64-key tiles ----------------
__global__ __launch_bounds__(256) void flash_attn(
    const _Float16* __restrict__ Qh,  // (B,T,DIM), pre-scaled by 1/sqrt(HDIM)
    const _Float16* __restrict__ Kh,  // (B,TKP,DIM), rows T..T+1 = tanh(g)*k_ad, pad zero
    const _Float16* __restrict__ Vt,  // (B,H,HDIM,TKP), pad zero
    _Float16* __restrict__ Oh) {      // (B,T,DIM)
  __shared__ _Float16 lp[8 * 16 * 64];  // per-wave P staging (C-layout -> A-layout), 16KB
  const int tid = threadIdx.x, lane = tid & 31, wave = tid >> 5;
  const int li = lane & 15, lh = lane >> 4;
  const int b = blockIdx.z, h = blockIdx.y;
  const int qr = blockIdx.x * 128 + wave * 16;

  // Q A-fragments: 16 rows x 128 dh, 4 chunks of K=32 (resident for whole kernel)
  v16h qf[4];
  {
    const _Float16* base = Qh + ((size_t)(b * TSEQ + qr + li)) * DIM + h * HDIM;
#pragma unroll
    for (int c = 0; c < 4; ++c) {
      const _Float16* p = base + c * 32 + lh * 8;
      qf[c] = cat16(*(const v8h*)p, *(const v8h*)(p + 16));
    }
  }

  const v8f vzero = {0.f, 0.f, 0.f, 0.f, 0.f, 0.f, 0.f, 0.f};
  v8f o[8];
#pragma unroll
  for (int n = 0; n < 8; ++n) o[n] = vzero;
  float mrow[8], lrow[8];
#pragma unroll
  for (int j = 0; j < 8; ++j) { mrow[j] = -3.0e38f; lrow[j] = 0.f; }

  _Float16* myP = &lp[wave * 16 * 64];

  for (int kb = 0; kb < TKP; kb += 64) {
    // S = Q K^T for 64 keys -> four 16x16 f32 accumulators
    v8f s[4];
#pragma unroll
    for (int nt = 0; nt < 4; ++nt) s[nt] = vzero;
    const _Float16* kbase = Kh + ((size_t)b * TKP) * DIM + h * HDIM;
#pragma unroll
    for (int c = 0; c < 4; ++c)
#pragma unroll
      for (int nt = 0; nt < 4; ++nt) {
        const _Float16* p = kbase + (size_t)(kb + nt * 16 + li) * DIM + c * 32 + lh * 8;
        v16h bfk = cat16(*(const v8h*)p, *(const v8h*)(p + 16));
        s[nt] = __builtin_amdgcn_wmma_f32_16x16x32_f16(
            false, qf[c], false, bfk, (short)0, s[nt], false, false);
      }
    // mask padded keys (last tile only)
    if (kb + 64 > KTOT) {
#pragma unroll
      for (int nt = 0; nt < 4; ++nt)
        if (kb + nt * 16 + li >= KTOT) {
#pragma unroll
          for (int v = 0; v < 8; ++v) s[nt][v] = -3.0e38f;
        }
    }
    // online softmax; row of element v = v + 8*lh; reductions across 16-lane groups.
    // Reduction + rescale cost amortizes over 64 keys (32 WMMAs) per tile.
    v8f scl;
#pragma unroll
    for (int j = 0; j < 8; ++j) {
      float tmax = fmaxf(fmaxf(s[0][j], s[1][j]), fmaxf(s[2][j], s[3][j]));
#pragma unroll
      for (int msk = 1; msk < 16; msk <<= 1) tmax = fmaxf(tmax, __shfl_xor(tmax, msk, 32));
      float mnew = fmaxf(mrow[j], tmax);
      float sc   = __expf(mrow[j] - mnew);
      float p0 = __expf(s[0][j] - mnew);
      float p1 = __expf(s[1][j] - mnew);
      float p2 = __expf(s[2][j] - mnew);
      float p3 = __expf(s[3][j] - mnew);
      float rsum = (p0 + p1) + (p2 + p3);
#pragma unroll
      for (int msk = 1; msk < 16; msk <<= 1) rsum += __shfl_xor(rsum, msk, 32);
      lrow[j] = lrow[j] * sc + rsum;
      mrow[j] = mnew;
      scl[j]  = sc;
      int r = j + 8 * lh;
      myP[r * 64 + li]      = (_Float16)p0;
      myP[r * 64 + 16 + li] = (_Float16)p1;
      myP[r * 64 + 32 + li] = (_Float16)p2;
      myP[r * 64 + 48 + li] = (_Float16)p3;
    }
#pragma unroll
    for (int n = 0; n < 8; ++n) o[n] *= scl;
    // P back as two A-fragments (wave-private LDS; DS ops in-order per wave)
    v16h pf[2];
#pragma unroll
    for (int ck = 0; ck < 2; ++ck) {
      const _Float16* p = &myP[li * 64 + ck * 32 + lh * 8];
      pf[ck] = cat16(*(const v8h*)p, *(const v8h*)(p + 16));
    }
    // O += P @ V  (Vt rows contiguous along keys -> perfect B-fragment loads)
    const _Float16* vb = Vt + (((size_t)b * NHEADS + h) * HDIM) * TKP;
#pragma unroll
    for (int n = 0; n < 8; ++n)
#pragma unroll
      for (int ck = 0; ck < 2; ++ck) {
        const _Float16* p = vb + (size_t)(n * 16 + li) * TKP + kb + ck * 32 + lh * 8;
        v16h bfv = cat16(*(const v8h*)p, *(const v8h*)(p + 16));
        o[n] = __builtin_amdgcn_wmma_f32_16x16x32_f16(
            false, pf[ck], false, bfv, (short)0, o[n], false, false);
      }
  }
  // normalize and store f16 (feeds output projection GEMM)
  float inv[8];
#pragma unroll
  for (int j = 0; j < 8; ++j) inv[j] = 1.f / lrow[j];
#pragma unroll
  for (int n = 0; n < 8; ++n)
#pragma unroll
    for (int v = 0; v < 8; ++v) {
      int row = qr + v + 8 * lh;
      int col = h * HDIM + n * 16 + li;
      Oh[((size_t)b * TSEQ + row) * DIM + col] = (_Float16)(o[n][v] * inv[v]);
    }
}

extern "C" void kernel_launch(void* const* d_in, const int* in_sizes, int n_in,
                              void* d_out, int out_size, void* d_ws, size_t ws_size,
                              hipStream_t stream) {
  const float* x  = (const float*)d_in[0];
  const float* hB = (const float*)d_in[1];
  const float* pB = (const float*)d_in[2];
  const float* Wq = (const float*)d_in[3];
  const float* bq = (const float*)d_in[4];
  const float* Wk = (const float*)d_in[5];
  const float* bk = (const float*)d_in[6];
  const float* Wv = (const float*)d_in[7];
  const float* bv = (const float*)d_in[8];
  const float* Wo = (const float*)d_in[9];
  const float* bo = (const float*)d_in[10];
  const float* g  = (const float*)d_in[11];
  float* out = (float*)d_out;
  (void)in_sizes; (void)n_in; (void)out_size; (void)ws_size;

  char* ws = (char*)d_ws;
  size_t off = 0;
  auto alloc = [&](size_t bytes) -> char* {
    char* p = ws + off;
    off += (bytes + 255) & ~(size_t)255;
    return p;
  };
  _Float16* xh  = (_Float16*)alloc((size_t)BATCH * TSEQ * DIM * 2);
  _Float16* Wqh = (_Float16*)alloc((size_t)DIM * DIM * 2);
  _Float16* Wkh = (_Float16*)alloc((size_t)DIM * DIM * 2);
  _Float16* Wvh = (_Float16*)alloc((size_t)DIM * DIM * 2);
  _Float16* Woh = (_Float16*)alloc((size_t)DIM * DIM * 2);
  _Float16* Qh  = (_Float16*)alloc((size_t)BATCH * TSEQ * DIM * 2);
  _Float16* Kh  = (_Float16*)alloc((size_t)BATCH * TKP * DIM * 2);
  _Float16* Vt  = (_Float16*)alloc((size_t)BATCH * NHEADS * HDIM * TKP * 2);
  _Float16* Oh  = (_Float16*)alloc((size_t)BATCH * TSEQ * DIM * 2);

  // zero padding regions (graph-capturable async memset)
  hipMemsetAsync(Kh, 0, (size_t)BATCH * TKP * DIM * 2, stream);
  hipMemsetAsync(Vt, 0, (size_t)BATCH * NHEADS * HDIM * TKP * 2, stream);

  const int nx = BATCH * TSEQ * DIM, nw = DIM * DIM;
  cvt_f16<<<(nx + 255) / 256, 256, 0, stream>>>(x, xh, nx);
  cvt_f16<<<(nw + 255) / 256, 256, 0, stream>>>(Wq, Wqh, nw);
  cvt_f16<<<(nw + 255) / 256, 256, 0, stream>>>(Wk, Wkh, nw);
  cvt_f16<<<(nw + 255) / 256, 256, 0, stream>>>(Wv, Wvh, nw);
  cvt_f16<<<(nw + 255) / 256, 256, 0, stream>>>(Wo, Woh, nw);

  const dim3 gg(DIM / 128, (BATCH * TSEQ) / 128);
  const float qscale = 0.08838834764831845f;  // 1/sqrt(HDIM), folded into Q
  gemm_xwT<0><<<gg, 256, 0, stream>>>(xh, Wqh, bq, Qh, BATCH * TSEQ, DIM, DIM, qscale, TSEQ, TSEQ);
  gemm_xwT<0><<<gg, 256, 0, stream>>>(xh, Wkh, bk, Kh, BATCH * TSEQ, DIM, DIM, 1.f, TSEQ, TKP);
  gemm_xwT<2><<<gg, 256, 0, stream>>>(xh, Wvh, bv, Vt, BATCH * TSEQ, DIM, DIM, 1.f, TSEQ, 0);

  const int na = BATCH * 2 * DIM;
  adapter_kv<<<(na + 255) / 256, 256, 0, stream>>>(hB, pB, Wk, bk, g, Kh, 0);
  adapter_kv<<<(na + 255) / 256, 256, 0, stream>>>(hB, pB, Wv, bv, g, Vt, 1);

  flash_attn<<<dim3(TSEQ / 128, NHEADS, BATCH), 256, 0, stream>>>(Qh, Kh, Vt, Oh);

  gemm_xwT<1><<<gg, 256, 0, stream>>>(Oh, Woh, bo, out, BATCH * TSEQ, DIM, DIM, 1.f, TSEQ, TSEQ);
}
